// AttentionPool2d_15925738733908
// MI455X (gfx1250) — matrix operand order repaired and checked by reading, not compiled, not run on hardware
//
#include <hip/hip_runtime.h>
#include <math.h>

// Shapes (fixed by the reference)
// B=64, S=32, D=256, H=8, DK=32, O=256, N=1025 tokens (CLS + 1024)
#define NB   64
#define ND   256
#define NH   8
#define NDK  32
#define NTOK 1025
#define NPAD 1056            // 33 * 32 : K-padded token count for WMMA
#define UP   260             // LDS pitch for u tile   (bank-conflict avoidance)
#define AP   1060            // LDS pitch for attn tile (bank-conflict avoidance)

typedef __attribute__((ext_vector_type(16))) _Float16 v16h;
typedef __attribute__((ext_vector_type(8)))  float    v8f;
typedef int v4i __attribute__((vector_size(16)));

#if __has_builtin(__builtin_amdgcn_global_load_async_to_lds_b128) && \
    __has_builtin(__builtin_amdgcn_s_wait_asynccnt)
#define HAVE_ASYNC 1
#else
#define HAVE_ASYNC 0
#endif

// builtin expects (addrspace(1) v4i* gsrc, addrspace(3) v4i* ldst, imm, imm)
#define GPTR(p) ((__attribute__((address_space(1))) v4i*)(float*)(p))
#define LPTR(p) ((__attribute__((address_space(3))) v4i*)(p))

// ---------------------------------------------------------------------------
// Kernel 1: partial sums over tokens for the per-batch mean (streams all of x)
// grid(64,8) block(256): block (b,chunk) reduces 128 tokens for all 256 dims.
// ---------------------------------------------------------------------------
__global__ __launch_bounds__(256) void k_mean(const float* __restrict__ x,
                                              float* __restrict__ partial) {
  const int b = blockIdx.x, chunk = blockIdx.y, d = threadIdx.x;
  const float* xb = x + ((size_t)b * 1024 + (size_t)chunk * 128) * ND;
  float s = 0.f;
  for (int m = 0; m < 128; ++m) {
    if (m + 16 < 128) __builtin_prefetch(&xb[(size_t)(m + 16) * ND + d], 0, 1);
    s += xb[(size_t)m * ND + d];
  }
  partial[((size_t)b * 8 + chunk) * ND + d] = s;
}

// ---------------------------------------------------------------------------
// Kernel 2: t_cls = mean + pos_emb[0];  q = t_cls*Wq + bq;
//           u[b,h,:] = scale * (Wk[:,h,:] @ q[b,h,:])   (heads padded to 16)
// grid(64) block(256)
// ---------------------------------------------------------------------------
__global__ __launch_bounds__(256) void k_qu(const float* __restrict__ partial,
                                            const float* __restrict__ pos,
                                            const float* __restrict__ Wq,
                                            const float* __restrict__ bq,
                                            const float* __restrict__ Wk,
                                            float* __restrict__ t_cls,
                                            float* __restrict__ u) {
  const int b = blockIdx.x, tid = threadIdx.x;
  __shared__ float tc[ND];
  __shared__ float q[ND];
  float s = 0.f;
  for (int c = 0; c < 8; ++c) s += partial[((size_t)b * 8 + c) * ND + tid];
  const float t = s * (1.0f / 1024.0f) + pos[tid];   // pos_emb[0]
  tc[tid] = t;
  t_cls[b * ND + tid] = t;
  __syncthreads();
  // q[h*32+dk] for tid = h*32+dk
  float acc = bq[tid];
  for (int d = 0; d < ND; ++d) acc += tc[d] * Wq[d * ND + tid];
  q[tid] = acc;
  __syncthreads();
  const float scale = 0.17677669529663689f;          // 1/sqrt(32)
  for (int i = tid; i < 16 * ND; i += 256) {
    const int h = i >> 8, d = i & 255;
    float uv = 0.f;
    if (h < NH) {
      for (int dk = 0; dk < NDK; ++dk)
        uv += Wk[d * ND + h * NDK + dk] * q[h * NDK + dk];
      uv *= scale;
    }
    u[((size_t)b * 16 + h) * ND + d] = uv;           // heads 8..15 zero
  }
}

// ---------------------------------------------------------------------------
// Kernel 3: scores[b,h,m] = T[m,:] . u[b,h,:]  via WMMA f32_16x16x32_f16
// u (16 heads x 256 d) is async-staged into LDS once per block and shared by
// all 8 waves. A = 16 tokens x 32 d-chunk of T (registers), 8 k-steps.
// grid(64,9) block(256): 1 wave = one 16-token tile (65 tiles).
// ---------------------------------------------------------------------------
__global__ __launch_bounds__(256) void k_scores(const float* __restrict__ x,
                                                const float* __restrict__ pos,
                                                const float* __restrict__ t_cls,
                                                const float* __restrict__ u,
                                                float* __restrict__ scores) {
  const int b = blockIdx.x;
  __shared__ float ulds[16 * UP];
  const float* ubase = u + (size_t)b * 16 * ND;
#if HAVE_ASYNC
  // 16 rows x 64 float4 = 1024 async B128 copies, 4 per thread
  for (int i = threadIdx.x; i < 1024; i += 256) {
    const int r = i >> 6, c4 = (i & 63) * 4;
    __builtin_amdgcn_global_load_async_to_lds_b128(
        GPTR(ubase + r * ND + c4), LPTR(ulds + r * UP + c4), 0, 0);
  }
  __builtin_amdgcn_s_wait_asynccnt(0);
#else
  for (int i = threadIdx.x; i < 16 * ND; i += 256)
    ulds[(i >> 8) * UP + (i & 255)] = ubase[i];
#endif
  __syncthreads();

  const int lane = threadIdx.x & 31;
  const int wave = threadIdx.x >> 5;
  const int tile = blockIdx.y * 8 + wave;
  if (tile >= 65) return;                 // wave-uniform: EXEC all-1 at WMMA
  const int mb  = tile * 16;
  const int l16 = lane & 15;
  const int hi  = lane >> 4;              // lane half selects K range
  // A operand: row = token mb+l16 of T (CLS row = t_cls, clamp past 1024)
  const int m  = mb + l16;
  const int mc = m > 1024 ? 1024 : m;
  const float* arow = (m == 0) ? (t_cls + (size_t)b * ND)
                               : (x + ((size_t)b * 1024 + (size_t)(mc - 1)) * ND);
  const float* prow = pos + (size_t)mc * ND;
  const float  pw   = (m == 0) ? 0.0f : 1.0f;   // pos[0] already inside t_cls
  // B operand: col = head l16 (from LDS)
  const float* urow = ulds + l16 * UP;
  const int ka = hi ? 16 : 0;             // A: K = ka..ka+15
  const int kb = hi ? 16 : 0;             // B: K = kb..kb+15

  v8f c = {};
  for (int dc = 0; dc < 8; ++dc) {        // 8 x 32 = 256 = D
    const int d0 = dc * 32;
    v16h a, bm;
#pragma unroll
    for (int e = 0; e < 8; ++e) {
      a[e]     = (_Float16)(arow[d0 + ka + e]      + pw * prow[d0 + ka + e]);
      a[8 + e] = (_Float16)(arow[d0 + ka + 8 + e] + pw * prow[d0 + ka + 8 + e]);
    }
#pragma unroll
    for (int e = 0; e < 16; ++e)
      bm[e] = (_Float16)urow[d0 + kb + e];
    c = __builtin_amdgcn_wmma_f32_16x16x32_f16(false, a, false, bm,
                                               (short)0, c, false, false);
  }
  // D layout: lane holds column h=l16, rows (hi*8 + j) -> tokens mb+hi*8+j
  float* srow = scores + ((size_t)b * 16 + l16) * NPAD + mb + hi * 8;
#pragma unroll
  for (int j = 0; j < 8; ++j) srow[j] = c[j];
}

// ---------------------------------------------------------------------------
// Kernel 4: softmax over 1025 tokens per (b,h); writes attn in place,
// zero-pads tokens 1025..1055 so WMMA K-padding is inert.
// grid(64) block(256): wave = head.
// ---------------------------------------------------------------------------
__global__ __launch_bounds__(256) void k_softmax(float* __restrict__ sc) {
  const int b = blockIdx.x;
  const int h = threadIdx.x >> 5;         // 0..7 (pad rows 8..15 stay zero)
  const int lane = threadIdx.x & 31;
  float* row = sc + ((size_t)b * 16 + h) * NPAD;
  float mx = -3.0e38f;
  for (int m = lane; m < NTOK; m += 32) mx = fmaxf(mx, row[m]);
  for (int off = 16; off > 0; off >>= 1) mx = fmaxf(mx, __shfl_xor(mx, off, 32));
  float sum = 0.f;
  for (int m = lane; m < NTOK; m += 32) sum += __expf(row[m] - mx);
  for (int off = 16; off > 0; off >>= 1) sum += __shfl_xor(sum, off, 32);
  const float inv = 1.0f / sum;
  for (int m = lane; m < NPAD; m += 32) {
    const float v = (m < NTOK) ? __expf(row[m] - mx) * inv : 0.0f;
    row[m] = v;
  }
}

// ---------------------------------------------------------------------------
// Kernel 5: w[b,h,:] = sum_m attn[b,h,m] * T[m,:]  via WMMA
// attn rows (16 x 1056) are identical for all 16 column tiles of a batch:
// async-stage them into LDS once per block. B = T columns (global), 33 steps.
// grid(64,2) block(256): wave = one 16-wide d column tile (16 tiles).
// ---------------------------------------------------------------------------
__global__ __launch_bounds__(256) void k_wsum(const float* __restrict__ x,
                                              const float* __restrict__ pos,
                                              const float* __restrict__ t_cls,
                                              const float* __restrict__ attn,
                                              float* __restrict__ w) {
  const int b = blockIdx.x;
  __shared__ float alds[16 * AP];                    // ~66 KB of the 320 KB LDS
  const float* abase = attn + (size_t)b * 16 * NPAD;
#if HAVE_ASYNC
  // 16 rows x 264 float4 = 4224 async B128 copies, ~17 per thread
  for (int i = threadIdx.x; i < 4224; i += 256) {
    const int r = i / 264, c4 = (i % 264) * 4;
    __builtin_amdgcn_global_load_async_to_lds_b128(
        GPTR(abase + r * NPAD + c4), LPTR(alds + r * AP + c4), 0, 0);
  }
  __builtin_amdgcn_s_wait_asynccnt(0);
#else
  for (int i = threadIdx.x; i < 16 * NPAD; i += 256)
    alds[(i / NPAD) * AP + (i % NPAD)] = abase[i];
#endif
  __syncthreads();

  const int lane = threadIdx.x & 31;
  const int wave = threadIdx.x >> 5;
  const int ct   = blockIdx.y * 8 + wave;          // 0..15
  const int l16  = lane & 15;
  const int hi   = lane >> 4;
  const int dcol = ct * 16 + l16;                  // B column: dim index
  const float* arow = alds + l16 * AP;             // A row: head l16 (LDS)
  const int ka = hi ? 16 : 0;
  const int kb = hi ? 16 : 0;

  v8f c = {};
  for (int kc = 0; kc < 33; ++kc) {                // 33 x 32 = 1056 tokens
    const int k0 = kc * 32;
    v16h a, bm;
#pragma unroll
    for (int e = 0; e < 8; ++e) {
      a[e]     = (_Float16)arow[k0 + ka + e];
      a[8 + e] = (_Float16)arow[k0 + ka + 8 + e];
    }
#pragma unroll
    for (int e = 0; e < 16; ++e) {
      const int m  = k0 + kb + e;                  // token index (K)
      const int mc = m > 1024 ? 1024 : m;          // attn==0 past 1024 anyway
      const float* trow = (m == 0) ? (t_cls + (size_t)b * ND)
                                   : (x + ((size_t)b * 1024 + (size_t)(mc - 1)) * ND);
      const float pv = (m == 0) ? 0.0f : pos[(size_t)mc * ND + dcol];
      bm[e] = (_Float16)(trow[dcol] + pv);
    }
    c = __builtin_amdgcn_wmma_f32_16x16x32_f16(false, a, false, bm,
                                               (short)0, c, false, false);
  }
  // D: lane holds column dcol, rows = heads hi*8+j (rows 8..15 unused)
#pragma unroll
  for (int j = 0; j < 8; ++j)
    w[((size_t)b * 16 + (hi * 8 + j)) * ND + dcol] = c[j];
}

// ---------------------------------------------------------------------------
// Kernel 6: ov = w @ Wv + bv ;  out = ov @ Wo + bo      (tiny, fp32 VALU)
// grid(64) block(256)
// ---------------------------------------------------------------------------
__global__ __launch_bounds__(256) void k_out(const float* __restrict__ w,
                                             const float* __restrict__ Wv,
                                             const float* __restrict__ bv,
                                             const float* __restrict__ Wo,
                                             const float* __restrict__ bo,
                                             float* __restrict__ out) {
  const int b = blockIdx.x, tid = threadIdx.x;
  const int h = tid >> 5;                      // tid = h*32 + dk
  __shared__ float ov[ND];
  const float* wb = w + ((size_t)b * 16 + h) * ND;
  float acc = bv[tid];                         // bv[h, dk]
  for (int d = 0; d < ND; ++d) acc += wb[d] * Wv[d * ND + tid];
  ov[tid] = acc;
  __syncthreads();
  float o = bo[tid];
  for (int i = 0; i < ND; ++i) o += ov[i] * Wo[i * ND + tid];
  out[(size_t)b * ND + tid] = o;
}

// ---------------------------------------------------------------------------
extern "C" void kernel_launch(void* const* d_in, const int* in_sizes, int n_in,
                              void* d_out, int out_size, void* d_ws, size_t ws_size,
                              hipStream_t stream) {
  (void)in_sizes; (void)n_in; (void)out_size; (void)ws_size;
  const float* x   = (const float*)d_in[0];
  const float* pos = (const float*)d_in[1];
  const float* Wq  = (const float*)d_in[2];
  const float* bq  = (const float*)d_in[3];
  const float* Wk  = (const float*)d_in[4];
  // d_in[5] = bk: q.bk is constant over tokens -> cancels in softmax
  const float* Wv  = (const float*)d_in[6];
  const float* bv  = (const float*)d_in[7];
  const float* Wo  = (const float*)d_in[8];
  const float* bo  = (const float*)d_in[9];
  float* out = (float*)d_out;

  // Workspace layout (floats): all chunks 16B-aligned
  float* ws      = (float*)d_ws;
  float* partial = ws;                       // 64*8*256    = 131072
  float* t_cls   = ws + 131072;              // 64*256      =  16384
  float* u       = ws + 147456;              // 64*16*256   = 262144
  float* sc      = ws + 409600;              // 64*16*1056  = 1081344 (scores/attn)
  float* w       = ws + 1490944;             // 64*16*256   = 262144
                                             // total ~6.7 MB

  k_mean   <<<dim3(NB, 8), 256, 0, stream>>>(x, partial);
  k_qu     <<<NB,          256, 0, stream>>>(partial, pos, Wq, bq, Wk, t_cls, u);
  k_scores <<<dim3(NB, 9), 256, 0, stream>>>(x, pos, t_cls, u, sc);
  k_softmax<<<NB,          256, 0, stream>>>(sc);
  k_wsum   <<<dim3(NB, 2), 256, 0, stream>>>(x, pos, t_cls, sc, w);
  k_out    <<<NB,          256, 0, stream>>>(w, Wv, bv, Wo, bo, out);
}
